// VMoEBlock_1967095022052
// MI455X (gfx1250) — matrix operand order repaired and checked by reading, not compiled, not run on hardware
//
#include <hip/hip_runtime.h>
#include <math.h>

#define Tt    65536
#define Dd    512
#define Hh    2048
#define Oo    512
#define Ee    8
#define CAP   8192
#define NBINS 4096
#define BM    128
#define BN    128
#define BK    32

typedef __attribute__((ext_vector_type(16))) _Float16 v16h;
typedef __attribute__((ext_vector_type(8)))  _Float16 v8h;
typedef __attribute__((ext_vector_type(4)))  _Float16 v4h;
typedef __attribute__((ext_vector_type(8)))  float    v8f;
typedef __attribute__((ext_vector_type(4)))  float    v4f;

// ---------------------------------------------------------------------------
// Kernel 1: fused LayerNorm + router (logits, noisy softmax, top-2) + weight
// histogram for capacity thresholding. One block (256 thr = 8 waves) per token.
// ---------------------------------------------------------------------------
__global__ __launch_bounds__(256) void ln_router(
    const float* __restrict__ x, const float* __restrict__ noise,
    const float* __restrict__ gamma, const float* __restrict__ beta,
    const float* __restrict__ Wg, const float* __restrict__ bg,
    float* __restrict__ xn, int* __restrict__ route_e,
    float* __restrict__ route_w, int* __restrict__ hist)
{
    __shared__ float xs[Dd];
    __shared__ float red[24];
    __shared__ float lg[Ee];
    const int t   = blockIdx.x;
    const int tid = threadIdx.x;
    const int w   = tid >> 5, l = tid & 31;

    const float* xr = x + (size_t)t * Dd;
    float v0 = xr[tid], v1 = xr[tid + 256];
    float s = v0 + v1, s2 = v0 * v0 + v1 * v1;
#pragma unroll
    for (int o = 16; o > 0; o >>= 1) {
        s  += __shfl_down(s,  o);
        s2 += __shfl_down(s2, o);
    }
    if (l == 0) { red[w] = s; red[8 + w] = s2; }
    __syncthreads();
    if (tid == 0) {
        float a = 0.f, b = 0.f;
#pragma unroll
        for (int i = 0; i < 8; i++) { a += red[i]; b += red[8 + i]; }
        float mu  = a / Dd;
        float var = b / Dd - mu * mu;
        red[16] = mu;
        red[17] = rsqrtf(var + 1e-5f);
    }
    __syncthreads();
    const float mu = red[16], rs = red[17];
    float n0 = (v0 - mu) * rs * gamma[tid]       + beta[tid];
    float n1 = (v1 - mu) * rs * gamma[tid + 256] + beta[tid + 256];
    xs[tid] = n0; xs[tid + 256] = n1;
    xn[(size_t)t * Dd + tid]       = n0;
    xn[(size_t)t * Dd + tid + 256] = n1;
    __syncthreads();

    // 8 waves -> 8 expert logits via strided dot + shuffle reduce
    {
        float acc = 0.f;
        const float* wg = Wg + (size_t)w * Dd;
        for (int k = l; k < Dd; k += 32) acc += xs[k] * wg[k];
#pragma unroll
        for (int o = 16; o > 0; o >>= 1) acc += __shfl_down(acc, o);
        if (l == 0) lg[w] = acc + bg[w];
    }
    __syncthreads();

    if (tid == 0) {
        float lo[Ee]; float mx = -1e30f;
#pragma unroll
        for (int e = 0; e < Ee; e++) {
            lo[e] = lg[e] + 0.125f * noise[(size_t)t * Ee + e];
            mx = fmaxf(mx, lo[e]);
        }
        float den = 0.f;
#pragma unroll
        for (int e = 0; e < Ee; e++) { lo[e] = __expf(lo[e] - mx); den += lo[e]; }
        const float inv = 1.f / den;
        int i0 = 0; float g0 = -1.f;
#pragma unroll
        for (int e = 0; e < Ee; e++) { float g = lo[e] * inv; if (g > g0) { g0 = g; i0 = e; } }
        int i1 = 0; float g1 = -1.f;
#pragma unroll
        for (int e = 0; e < Ee; e++) { if (e == i0) continue; float g = lo[e] * inv; if (g > g1) { g1 = g; i1 = e; } }
        const float ws = g0 + g1 + 1e-20f;
        const float w0 = g0 / ws, w1 = g1 / ws;
        route_e[2 * t]     = i0; route_w[2 * t]     = w0;
        route_e[2 * t + 1] = i1; route_w[2 * t + 1] = w1;
        int b0 = min(NBINS - 1, (int)(w0 * NBINS));
        int b1 = min(NBINS - 1, (int)(w1 * NBINS));
        atomicAdd(&hist[i0 * NBINS + b0], 1);
        atomicAdd(&hist[i1 * NBINS + b1], 1);
    }
}

// ---------------------------------------------------------------------------
// Kernel 2: per-expert gate-weight admission threshold from histogram.
// ---------------------------------------------------------------------------
__global__ void find_thresh(const int* __restrict__ hist,
                            int* __restrict__ thr_bin, int* __restrict__ thr_allow)
{
    const int e = threadIdx.x;
    if (e >= Ee) return;
    long cum = 0; int tb = -1, allow = 0;
    for (int b = NBINS - 1; b >= 0; b--) {
        int c = hist[e * NBINS + b];
        if (cum + c >= CAP) { tb = b; allow = CAP - (int)cum; break; }
        cum += c;
    }
    thr_bin[e] = tb; thr_allow[e] = allow;
}

// ---------------------------------------------------------------------------
// Kernel 3: slot dispatch into per-expert capacity buffers.
// ---------------------------------------------------------------------------
__global__ __launch_bounds__(256) void dispatch(
    const int* __restrict__ route_e, const float* __restrict__ route_w,
    const int* __restrict__ thr_bin, const int* __restrict__ thr_allow,
    int* __restrict__ tcnt, int* __restrict__ cnt,
    int* __restrict__ slot_tok, float* __restrict__ slot_w,
    int* __restrict__ processed)
{
    const int s = blockIdx.x * 256 + threadIdx.x;
    if (s >= 2 * Tt) return;
    const int   e = route_e[s];
    const float w = route_w[s];
    const int   b = min(NBINS - 1, (int)(w * NBINS));
    const int  tb = thr_bin[e];
    bool admit = b > tb;
    if (!admit && b == tb) {
        int k = atomicAdd(&tcnt[e], 1);
        admit = k < thr_allow[e];
    }
    if (admit) {
        int pos = atomicAdd(&cnt[e], 1);
        if (pos < CAP) {
            slot_tok[e * CAP + pos] = s >> 1;
            slot_w[e * CAP + pos]   = w;
            processed[s >> 1]       = 1;
        }
    }
}

// ---------------------------------------------------------------------------
// Kernel 4: expert GEMM1 (gathered xn[cap,512] x W1[512,2048]) + GELU -> f16 h.
// 128x128 block tile, 8 waves of 32x64 (2x4 wmma frags), K-step 32.
// Software-pipelined: tile k+1 global->regs issued before the WMMA block of
// tile k, so HBM/L2 latency overlaps matrix-core work.
// ---------------------------------------------------------------------------
__global__ __launch_bounds__(256) void gemm1(
    const float* __restrict__ xn, const int* __restrict__ toks,
    const float* __restrict__ W1e, const float* __restrict__ b1e,
    _Float16* __restrict__ hbuf)
{
    __shared__ __align__(16) _Float16 As[BM * BK];   // [m][k]
    __shared__ __align__(32) _Float16 Bs[BN * BK];   // [n][k] transposed
    const int tid = threadIdx.x;
    const int m0 = blockIdx.x * BM, n0 = blockIdx.y * BN;
    const int wave = tid >> 5, lane = tid & 31;
    const int wm = wave >> 1, wn = wave & 1;   // 4 x 2 wave grid
    const int g = lane >> 4, ln = lane & 15;

    // loop-invariant gather setup (branch-free in the K loop)
    const int   ar   = tid >> 1;                 // A row this thread stages
    const int   akh  = (tid & 1) * 16;           // which 16-k half
    const int   tk   = toks[m0 + ar];
    const float amask = (tk >= 0) ? 1.f : 0.f;
    const float* arow = xn + (size_t)(tk >= 0 ? tk : 0) * Dd;
    const int   bkb = tid >> 5;                  // B k-block (0..7) -> 4 k rows
    const int   bnb = tid & 31;                  // B n-block (0..31) -> 4 n cols

    v8f acc[2][4];
#pragma unroll
    for (int i = 0; i < 2; i++)
#pragma unroll
        for (int j = 0; j < 4; j++)
#pragma unroll
            for (int r = 0; r < 8; r++) acc[i][j][r] = 0.f;

    v4f at[4], bt[4];
#pragma unroll
    for (int q = 0; q < 4; q++)
        at[q] = *(const v4f*)(arow + akh + 4 * q);
#pragma unroll
    for (int dk = 0; dk < 4; dk++)
        bt[dk] = *(const v4f*)&W1e[(size_t)(bkb * 4 + dk) * Hh + n0 + bnb * 4];

    for (int k0 = 0; k0 < Dd; k0 += BK) {
        __syncthreads();
        // ---- commit current tile regs -> LDS (f32 -> f16)
        v8h p0, p1;
#pragma unroll
        for (int q = 0; q < 4; q++)
#pragma unroll
            for (int c = 0; c < 4; c++) {
                _Float16 hv = (_Float16)(at[q][c] * amask);
                if (q < 2) p0[q * 4 + c] = hv; else p1[(q - 2) * 4 + c] = hv;
            }
        *(v8h*)&As[ar * BK + akh]     = p0;
        *(v8h*)&As[ar * BK + akh + 8] = p1;
#pragma unroll
        for (int dn = 0; dn < 4; dn++) {
            v4h pk = { (_Float16)bt[0][dn], (_Float16)bt[1][dn],
                       (_Float16)bt[2][dn], (_Float16)bt[3][dn] };
            *(v4h*)&Bs[(bnb * 4 + dn) * BK + bkb * 4] = pk;
        }
        __syncthreads();

        // ---- prefetch next tile into regs (overlaps with WMMA below)
        if (k0 + BK < Dd) {
            const int kn = k0 + BK;
#pragma unroll
            for (int q = 0; q < 4; q++)
                at[q] = *(const v4f*)(arow + kn + akh + 4 * q);
#pragma unroll
            for (int dk = 0; dk < 4; dk++)
                bt[dk] = *(const v4f*)&W1e[(size_t)(kn + bkb * 4 + dk) * Hh + n0 + bnb * 4];
        }

        // ---- fragments + WMMA
        v16h a[2], b[4];
#pragma unroll
        for (int i = 0; i < 2; i++) {         // A frag: 16 contiguous K halves
            const _Float16* ap = &As[(wm * 32 + i * 16 + ln) * BK + g * 16];
            v8h lo = *(const v8h*)ap;
            v8h hi = *(const v8h*)(ap + 8);
#pragma unroll
            for (int q = 0; q < 8; q++) { a[i][q] = lo[q]; a[i][8 + q] = hi[q]; }
        }
#pragma unroll
        for (int i = 0; i < 4; i++) {         // B frag: 16 contiguous K halves
            const _Float16* bp = &Bs[(wn * 64 + i * 16 + ln) * BK + g * 16];
            b[i] = *(const v16h*)bp;
        }
#pragma unroll
        for (int i = 0; i < 2; i++)
#pragma unroll
            for (int j = 0; j < 4; j++)
                acc[i][j] = __builtin_amdgcn_wmma_f32_16x16x32_f16(
                    false, a[i], false, b[j], (short)0, acc[i][j], false, false);
    }

    // epilogue: bias + exact GELU, store f16
#pragma unroll
    for (int i = 0; i < 2; i++)
#pragma unroll
        for (int j = 0; j < 4; j++) {
            int nG = n0 + wn * 64 + j * 16 + ln;
            float bias = b1e[nG];
            int mBase = m0 + wm * 32 + i * 16 + g * 8;
#pragma unroll
            for (int r = 0; r < 8; r++) {
                float v = acc[i][j][r] + bias;
                float ge = 0.5f * v * (1.f + erff(v * 0.70710678f));
                hbuf[(size_t)(mBase + r) * Hh + nG] = (_Float16)ge;
            }
        }
}

// ---------------------------------------------------------------------------
// Kernel 5: expert GEMM2 (h[cap,2048] x W2[2048,512]) + bias, gate-weighted
// atomic scatter-add into outputs. Same software-pipelined structure.
// ---------------------------------------------------------------------------
__global__ __launch_bounds__(256) void gemm2(
    const _Float16* __restrict__ hbuf, const float* __restrict__ W2e,
    const float* __restrict__ b2e, const int* __restrict__ toks,
    const float* __restrict__ wts, float* __restrict__ out)
{
    __shared__ __align__(16) _Float16 As[BM * BK];
    __shared__ __align__(32) _Float16 Bs[BN * BK];
    __shared__ int   tok_s[BM];
    __shared__ float w_s[BM];
    const int tid = threadIdx.x;
    const int m0 = blockIdx.x * BM, n0 = blockIdx.y * BN;
    if (tid < BM) { tok_s[tid] = toks[m0 + tid]; w_s[tid] = wts[m0 + tid]; }
    const int wave = tid >> 5, lane = tid & 31;
    const int wm = wave >> 1, wn = wave & 1;
    const int g = lane >> 4, ln = lane & 15;

    const int ar  = tid >> 1;
    const int akh = (tid & 1) * 16;
    const _Float16* arow = hbuf + (size_t)(m0 + ar) * Hh;
    const int bkb = tid >> 5;
    const int bnb = tid & 31;

    v8f acc[2][4];
#pragma unroll
    for (int i = 0; i < 2; i++)
#pragma unroll
        for (int j = 0; j < 4; j++)
#pragma unroll
            for (int r = 0; r < 8; r++) acc[i][j][r] = 0.f;

    v8h p0 = *(const v8h*)(arow + akh);
    v8h p1 = *(const v8h*)(arow + akh + 8);
    v4f bt[4];
#pragma unroll
    for (int dk = 0; dk < 4; dk++)
        bt[dk] = *(const v4f*)&W2e[(size_t)(bkb * 4 + dk) * Oo + n0 + bnb * 4];

    for (int k0 = 0; k0 < Hh; k0 += BK) {
        __syncthreads();
        // ---- commit current tile regs -> LDS
        *(v8h*)&As[ar * BK + akh]     = p0;
        *(v8h*)&As[ar * BK + akh + 8] = p1;
#pragma unroll
        for (int dn = 0; dn < 4; dn++) {
            v4h pk = { (_Float16)bt[0][dn], (_Float16)bt[1][dn],
                       (_Float16)bt[2][dn], (_Float16)bt[3][dn] };
            *(v4h*)&Bs[(bnb * 4 + dn) * BK + bkb * 4] = pk;
        }
        __syncthreads();

        // ---- prefetch next tile into regs
        if (k0 + BK < Hh) {
            const int kn = k0 + BK;
            p0 = *(const v8h*)(arow + kn + akh);
            p1 = *(const v8h*)(arow + kn + akh + 8);
#pragma unroll
            for (int dk = 0; dk < 4; dk++)
                bt[dk] = *(const v4f*)&W2e[(size_t)(kn + bkb * 4 + dk) * Oo + n0 + bnb * 4];
        }

        v16h a[2], b[4];
#pragma unroll
        for (int i = 0; i < 2; i++) {
            const _Float16* ap = &As[(wm * 32 + i * 16 + ln) * BK + g * 16];
            v8h lo = *(const v8h*)ap;
            v8h hi = *(const v8h*)(ap + 8);
#pragma unroll
            for (int q = 0; q < 8; q++) { a[i][q] = lo[q]; a[i][8 + q] = hi[q]; }
        }
#pragma unroll
        for (int i = 0; i < 4; i++) {
            const _Float16* bp = &Bs[(wn * 64 + i * 16 + ln) * BK + g * 16];
            b[i] = *(const v16h*)bp;
        }
#pragma unroll
        for (int i = 0; i < 2; i++)
#pragma unroll
            for (int j = 0; j < 4; j++)
                acc[i][j] = __builtin_amdgcn_wmma_f32_16x16x32_f16(
                    false, a[i], false, b[j], (short)0, acc[i][j], false, false);
    }

#pragma unroll
    for (int i = 0; i < 2; i++)
#pragma unroll
        for (int j = 0; j < 4; j++) {
            int nG = n0 + wn * 64 + j * 16 + ln;
            float bias = b2e[nG];
            int mBase = wm * 32 + i * 16 + g * 8;   // row within block tile
#pragma unroll
            for (int r = 0; r < 8; r++) {
                int mr  = mBase + r;
                int tk  = tok_s[mr];
                if (tk >= 0) {
                    float v = (acc[i][j][r] + bias) * w_s[mr];
                    atomicAdd(&out[(size_t)tk * Oo + nG], v);
                }
            }
        }
}

// ---------------------------------------------------------------------------
// Kernel 6: passthrough of normed x for unprocessed tokens (O == D == 512).
// ---------------------------------------------------------------------------
__global__ __launch_bounds__(256) void passthrough(
    const int* __restrict__ processed, const float* __restrict__ xn,
    float* __restrict__ out)
{
    size_t idx = (size_t)blockIdx.x * 256 + threadIdx.x;
    int t = (int)(idx >> 9);
    if (!processed[t]) out[idx] += xn[idx];
}

// ---------------------------------------------------------------------------
extern "C" void kernel_launch(void* const* d_in, const int* in_sizes, int n_in,
                              void* d_out, int out_size, void* d_ws, size_t ws_size,
                              hipStream_t stream)
{
    const float* x     = (const float*)d_in[0];
    const float* noise = (const float*)d_in[1];
    const float* gamma = (const float*)d_in[2];
    const float* beta  = (const float*)d_in[3];
    const float* Wg    = (const float*)d_in[4];
    const float* bg    = (const float*)d_in[5];
    const float* W1    = (const float*)d_in[6];
    const float* b1    = (const float*)d_in[7];
    const float* W2    = (const float*)d_in[8];
    const float* b2    = (const float*)d_in[9];
    float* out = (float*)d_out;

    char* ws = (char*)d_ws;
    size_t off = 0;
    float* xn        = (float*)(ws + off); off += (size_t)Tt * Dd * 4;
    int*   route_e   = (int*)  (ws + off); off += (size_t)Tt * 2 * 4;
    float* route_w   = (float*)(ws + off); off += (size_t)Tt * 2 * 4;
    int*   hist      = (int*)  (ws + off); off += (size_t)Ee * NBINS * 4;
    int*   tcnt      = (int*)  (ws + off); off += Ee * 4;
    int*   cnt       = (int*)  (ws + off); off += Ee * 4;
    int*   thr_bin   = (int*)  (ws + off); off += Ee * 4;
    int*   thr_allow = (int*)  (ws + off); off += Ee * 4;
    int*   slot_tok  = (int*)  (ws + off); off += (size_t)Ee * CAP * 4;
    float* slot_w    = (float*)(ws + off); off += (size_t)Ee * CAP * 4;
    int*   processed = (int*)  (ws + off); off += (size_t)Tt * 4;
    _Float16* hbuf   = (_Float16*)(ws + off); off += (size_t)CAP * Hh * 2;

    // init (graph-capture-safe async memsets): outputs + aux_loss = 0,
    // histogram/counters = 0, slot tokens = -1, processed = 0.
    hipMemsetAsync(out, 0, (size_t)out_size * sizeof(float), stream);
    hipMemsetAsync(hist, 0, (size_t)Ee * NBINS * 4 + 2 * Ee * 4, stream); // hist+tcnt+cnt
    hipMemsetAsync(slot_tok, 0xFF, (size_t)Ee * CAP * 4, stream);
    hipMemsetAsync(slot_w, 0, (size_t)Ee * CAP * 4, stream);
    hipMemsetAsync(processed, 0, (size_t)Tt * 4, stream);

    ln_router<<<Tt, 256, 0, stream>>>(x, noise, gamma, beta, Wg, bg,
                                      xn, route_e, route_w, hist);
    find_thresh<<<1, Ee, 0, stream>>>(hist, thr_bin, thr_allow);
    dispatch<<<(2 * Tt) / 256, 256, 0, stream>>>(route_e, route_w, thr_bin, thr_allow,
                                                 tcnt, cnt, slot_tok, slot_w, processed);
    for (int e = 0; e < Ee; e++) {
        gemm1<<<dim3(CAP / BM, Hh / BN), 256, 0, stream>>>(
            xn, slot_tok + (size_t)e * CAP,
            W1 + (size_t)e * Dd * Hh, b1 + (size_t)e * Hh, hbuf);
        gemm2<<<dim3(CAP / BM, Oo / BN), 256, 0, stream>>>(
            hbuf, W2 + (size_t)e * Hh * Oo, b2 + (size_t)e * Oo,
            slot_tok + (size_t)e * CAP, slot_w + (size_t)e * CAP, out);
    }
    passthrough<<<((size_t)Tt * Dd) / 256, 256, 0, stream>>>(processed, xn, out);
}
